// DeformableGatedFusion_59150289600929
// MI455X (gfx1250) — matrix operand order repaired and checked by reading, not compiled, not run on hardware
//
#include <hip/hip_runtime.h>
#include <math.h>

// ---------------------------------------------------------------------------
// DeformableGatedFusion for MI455X (gfx1250, wave32, WMMA).
//
// Dense GEMMs run on v_wmma_f32_16x16x32_bf16 with a bf16 hi/lo error-
// compensated split (3 WMMAs per 16x16x32 step) for ~fp32 accuracy.
// The f32->bf16 split is done ONCE per tile at LDS-staging time; LDS holds
// pre-packed bf16-pair planes so fragment building is pure ds_load_b32.
// Workload is HBM-bound (~105 GFLOP vs ~470MB traffic @ 23.3 TB/s).
// ---------------------------------------------------------------------------

typedef __attribute__((ext_vector_type(16))) __bf16        v16bf;
typedef __attribute__((ext_vector_type(8)))  float         v8f;
typedef __attribute__((ext_vector_type(8)))  unsigned int  v8u;

#define D_MODEL 384
#define NQ      4096
#define BATCH   8
#define NCUT    730
#define DCUT    1024

// ---------------------------------------------------------------------------
// bf16 split/pack helpers
// ---------------------------------------------------------------------------
__device__ __forceinline__ unsigned pack2(__bf16 a, __bf16 b) {
  unsigned short ua = __builtin_bit_cast(unsigned short, a);
  unsigned short ub = __builtin_bit_cast(unsigned short, b);
  return ((unsigned)ub << 16) | (unsigned)ua;
}

// split x0,x1 (adjacent K) into packed hi-pair and lo-pair words
__device__ __forceinline__ void split_pair(float x0, float x1,
                                           unsigned& hi, unsigned& lo) {
  __bf16 h0 = (__bf16)x0;
  __bf16 h1 = (__bf16)x1;
  __bf16 l0 = (__bf16)(x0 - (float)h0);
  __bf16 l1 = (__bf16)(x1 - (float)h1);
  hi = pack2(h0, h1);
  lo = pack2(l0, l1);
}

// ---------------------------------------------------------------------------
// WMMA fragment loads from pre-packed LDS planes (contiguous K order, same for A and B, wave32)
// A 16x32: lane L -> m = L%16; VGPR v holds K pair k0 = (L/16)*16 + 2v
//   -> plane word [m][(L/16)*8 + v]
// B 32x16: lane L -> n = L%16; VGPR v holds K pair k0 = (L/16)*16 + 2v
//   -> plane word [n][(L/16)*8 + v]
// ---------------------------------------------------------------------------
__device__ __forceinline__ v16bf frag_A(const unsigned (*P)[17], int m, int hl) {
  v8u r;
#pragma unroll
  for (int v = 0; v < 8; ++v) {
    r[v] = P[m][hl * 8 + v];
  }
  return __builtin_bit_cast(v16bf, r);
}

__device__ __forceinline__ v16bf frag_B(const unsigned (*P)[17], int n, int hl) {
  v8u r;
#pragma unroll
  for (int v = 0; v < 8; ++v) r[v] = P[n][hl * 8 + v];
  return __builtin_bit_cast(v16bf, r);
}

// Error-compensated bf16 product: ~fp32 result with 3 WMMAs.
__device__ __forceinline__ v8f wmma3(v16bf ah, v16bf al, v16bf bh, v16bf bl, v8f c) {
  c = __builtin_amdgcn_wmma_f32_16x16x32_bf16(false, ah, false, bh, (short)0, c, false, false);
  c = __builtin_amdgcn_wmma_f32_16x16x32_bf16(false, ah, false, bl, (short)0, c, false, false);
  c = __builtin_amdgcn_wmma_f32_16x16x32_bf16(false, al, false, bh, (short)0, c, false, false);
  return c;
}

template <int ACT>  // 0=none, 1=gelu(exact), 2=relu
__device__ __forceinline__ float activate(float x) {
  if (ACT == 1) return 0.5f * x * (1.0f + erff(x * 0.70710678118654752f));
  if (ACT == 2) return fmaxf(x, 0.0f);
  return x;
}

// ---------------------------------------------------------------------------
// Generic GEMM: Y[M,N] = act(X[M,K] + bias) ; X[M,K] @ W[K,N]
// 256 threads = 8 waves, block tile 128x128, wave tile 32x64, K step 32.
// LDS: packed bf16-pair planes, split done once per element at staging.
// ---------------------------------------------------------------------------
template <int ACT>
__global__ __launch_bounds__(256) void gemm_wmma(
    const float* __restrict__ X, const float* __restrict__ W,
    const float* __restrict__ bias, float* __restrict__ Y,
    int M, int N, int K) {
  __shared__ unsigned Ah[128][17], Al[128][17];  // [m][kpair]
  __shared__ unsigned Bh[128][17], Bl[128][17];  // [n][kpair] (transposed W)

  const int tid  = threadIdx.x;
  const int lane = tid & 31;
  const int wave = tid >> 5;
  const int wm   = wave >> 1;  // 0..3 -> M sub-tile
  const int wn   = wave & 1;   // 0..1 -> N sub-tile
  const int hl   = lane >> 4;  // lane half
  const int l16  = lane & 15;
  const int bm   = blockIdx.y * 128;
  const int bn   = blockIdx.x * 128;

  v8f acc[2][4];
#pragma unroll
  for (int mt = 0; mt < 2; ++mt)
#pragma unroll
    for (int nt = 0; nt < 4; ++nt)
#pragma unroll
      for (int i = 0; i < 8; ++i) acc[mt][nt][i] = 0.0f;

  for (int kt = 0; kt < K; kt += 32) {
    // --- stage X tile: float4 along K, split+pack once per element ---
#pragma unroll
    for (int it = 0; it < 4; ++it) {
      int row = (tid >> 3) + it * 32;
      int col = (tid & 7) * 4;  // multiple of 4
      int gm  = bm + row;
      float4 v = make_float4(0.f, 0.f, 0.f, 0.f);
      if (gm < M) {
        v = *reinterpret_cast<const float4*>(X + (size_t)gm * K + kt + col);
        if (kt + 32 < K)  // prefetch next K tile (global_prefetch_b8)
          __builtin_prefetch(X + (size_t)gm * K + kt + 32 + col, 0, 1);
      }
      unsigned h0, l0, h1, l1;
      split_pair(v.x, v.y, h0, l0);
      split_pair(v.z, v.w, h1, l1);
      int kp = col >> 1;
      Ah[row][kp]     = h0;  Al[row][kp]     = l0;
      Ah[row][kp + 1] = h1;  Al[row][kp + 1] = l1;
    }
    // --- stage W tile transposed: each thread owns 2 cols x 2 adjacent K ---
    {
      const int nIdx = tid & 63;   // column pair id (n = 2*nIdx, 2*nIdx+1)
      const int kp0  = tid >> 6;   // 0..3
      const int gn0  = bn + nIdx * 2;
      const bool ok0 = (gn0 + 0) < N;
      const bool ok1 = (gn0 + 1) < N;
#pragma unroll
      for (int it = 0; it < 4; ++it) {
        int kp = kp0 + it * 4;     // k-pair 0..15
        const float* w0 = W + (size_t)(kt + kp * 2) * N + gn0;      // k even
        const float* w1 = w0 + N;                                    // k odd
        float a0 = ok0 ? w0[0] : 0.f;
        float b0 = ok1 ? w0[1] : 0.f;
        float a1 = ok0 ? w1[0] : 0.f;
        float b1 = ok1 ? w1[1] : 0.f;
        unsigned h, l;
        split_pair(a0, a1, h, l);
        Bh[nIdx * 2][kp] = h;  Bl[nIdx * 2][kp] = l;
        split_pair(b0, b1, h, l);
        Bh[nIdx * 2 + 1][kp] = h;  Bl[nIdx * 2 + 1][kp] = l;
      }
    }
    __syncthreads();

    // --- fragments (pure ds_load_b32) + WMMA ---
    v16bf fAh[2], fAl[2];
#pragma unroll
    for (int mt = 0; mt < 2; ++mt) {
      int m = wm * 32 + mt * 16 + l16;
      fAh[mt] = frag_A(Ah, m, hl);
      fAl[mt] = frag_A(Al, m, hl);
    }
#pragma unroll
    for (int nt = 0; nt < 4; ++nt) {
      int n = wn * 64 + nt * 16 + l16;
      v16bf fBh = frag_B(Bh, n, hl);
      v16bf fBl = frag_B(Bl, n, hl);
#pragma unroll
      for (int mt = 0; mt < 2; ++mt)
        acc[mt][nt] = wmma3(fAh[mt], fAl[mt], fBh, fBl, acc[mt][nt]);
    }
    __syncthreads();
  }

  // --- epilogue: bias + activation, guarded store ---
#pragma unroll
  for (int nt = 0; nt < 4; ++nt) {
    int gn = bn + wn * 64 + nt * 16 + l16;
    float bv = (gn < N) ? bias[gn] : 0.0f;
#pragma unroll
    for (int mt = 0; mt < 2; ++mt) {
#pragma unroll
      for (int r = 0; r < 8; ++r) {
        int gm = bm + wm * 32 + mt * 16 + r + hl * 8;
        if (gm < M && gn < N) {
          float v = activate<ACT>(acc[mt][nt][r] + bv);
          Y[(size_t)gm * N + gn] = v;
        }
      }
    }
  }
}

// ---------------------------------------------------------------------------
// LayerNorm over rows of length `cols` (cut3r: cols=1024). One block per row.
// ---------------------------------------------------------------------------
__global__ __launch_bounds__(256) void ln_rows_kernel(
    const float* __restrict__ X, const float* __restrict__ g,
    const float* __restrict__ b, float* __restrict__ Y, int cols) {
  __shared__ float rs[256], rq[256];
  const int tid = threadIdx.x;
  const float* x = X + (size_t)blockIdx.x * cols;
  float s = 0.f, ss = 0.f;
  for (int c = tid; c < cols; c += 256) {
    float v = x[c];
    s += v; ss += v * v;
  }
  rs[tid] = s; rq[tid] = ss;
  __syncthreads();
  for (int st = 128; st > 0; st >>= 1) {
    if (tid < st) { rs[tid] += rs[tid + st]; rq[tid] += rq[tid + st]; }
    __syncthreads();
  }
  float mean = rs[0] / cols;
  float var  = rq[0] / cols - mean * mean;
  float rstd = rsqrtf(var + 1e-5f);
  float* y = Y + (size_t)blockIdx.x * cols;
  for (int c = tid; c < cols; c += 256)
    y[c] = (x[c] - mean) * rstd * g[c] + b[c];
}

// ---------------------------------------------------------------------------
// Visual LN + transpose materialization:
//   vis [B, 384, 4096] -> vT [B,4096,384] (raw) and vn [B,4096,384] (LN'd)
// Block handles 32 queries x 384 channels via LDS transpose.
// ---------------------------------------------------------------------------
__global__ __launch_bounds__(256) void ln_visual_kernel(
    const float* __restrict__ vis, const float* __restrict__ g,
    const float* __restrict__ b, float* __restrict__ vT, float* __restrict__ vn) {
  __shared__ float S[D_MODEL][33];
  __shared__ float redS[32][8], redQ[32][8];
  __shared__ float stat[32][2];
  const int tid = threadIdx.x;
  const int bq0 = blockIdx.x * 32;
  const int bb  = blockIdx.y;
  const float* base = vis + (size_t)bb * D_MODEL * NQ;

  const int qi = tid & 31;
  for (int c = tid >> 5; c < D_MODEL; c += 8)
    S[c][qi] = base[(size_t)c * NQ + bq0 + qi];
  __syncthreads();

  const int ql = tid >> 3, part = tid & 7;
  float s = 0.f, ss = 0.f;
  for (int c = part; c < D_MODEL; c += 8) {
    float v = S[c][ql];
    s += v; ss += v * v;
  }
  redS[ql][part] = s; redQ[ql][part] = ss;
  __syncthreads();
  if (part == 0) {
    float ts = 0.f, tq = 0.f;
#pragma unroll
    for (int p = 0; p < 8; ++p) { ts += redS[ql][p]; tq += redQ[ql][p]; }
    float mean = ts / D_MODEL;
    float var  = tq / D_MODEL - mean * mean;
    stat[ql][0] = mean;
    stat[ql][1] = rsqrtf(var + 1e-5f);
  }
  __syncthreads();

  for (int idx = tid; idx < 32 * D_MODEL; idx += 256) {
    int c = idx % D_MODEL, q = idx / D_MODEL;  // c fastest -> coalesced stores
    float v = S[c][q];
    size_t o = ((size_t)bb * NQ + bq0 + q) * D_MODEL + c;
    vT[o] = v;
    vn[o] = (v - stat[q][0]) * stat[q][1] * g[c] + b[c];
  }
}

// ---------------------------------------------------------------------------
// Deformable bilinear sampling. One 128-thread block per (b, q).
// ---------------------------------------------------------------------------
__global__ __launch_bounds__(128) void sample_kernel(
    const float* __restrict__ offr, const float* __restrict__ attr,
    const float* __restrict__ cp, float* __restrict__ sout) {
  __shared__ float loc[64];
  __shared__ float aw[32];
  const int t  = threadIdx.x;
  const int q  = blockIdx.x;
  const int bb = blockIdx.y;
  const size_t bq = (size_t)bb * NQ + q;

  if (t < 64) {
    float o = offr[bq * 64 + t];
    loc[t] = 2.0f / (1.0f + expf(-o)) - 1.0f;  // 2*sigmoid - 1
  } else if (t < 96) {
    aw[t - 64] = attr[bq * 32 + (t - 64)];
  }
  __syncthreads();
  if (t < 8) {  // softmax over 4 points per head
    float v0 = aw[t * 4 + 0], v1 = aw[t * 4 + 1], v2 = aw[t * 4 + 2], v3 = aw[t * 4 + 3];
    float m = fmaxf(fmaxf(v0, v1), fmaxf(v2, v3));
    float e0 = expf(v0 - m), e1 = expf(v1 - m), e2 = expf(v2 - m), e3 = expf(v3 - m);
    float inv = 1.0f / (e0 + e1 + e2 + e3);
    aw[t * 4 + 0] = e0 * inv; aw[t * 4 + 1] = e1 * inv;
    aw[t * 4 + 2] = e2 * inv; aw[t * 4 + 3] = e3 * inv;
  }
  __syncthreads();

  const float* cpb = cp + ((size_t)bb * NCUT + 1) * D_MODEL;  // skip CLS token
  for (int c = t; c < D_MODEL; c += 128) {
    int h = c / 48;
    float acc = 0.f;
#pragma unroll
    for (int p = 0; p < 4; ++p) {
      float x = loc[h * 8 + p * 2 + 0];
      float y = loc[h * 8 + p * 2 + 1];
      float ix = ((x + 1.0f) * 27.0f - 1.0f) * 0.5f;
      float iy = ((y + 1.0f) * 27.0f - 1.0f) * 0.5f;
      float x0f = floorf(ix), y0f = floorf(iy);
      float wx1 = ix - x0f, wx0 = 1.0f - wx1;
      float wy1 = iy - y0f, wy0 = 1.0f - wy1;
      int x0 = (int)x0f, y0 = (int)y0f;
      float v00 = 0.f, v10 = 0.f, v01 = 0.f, v11 = 0.f;
      bool xa = (x0 >= 0) && (x0 < 27), xb = (x0 + 1 >= 0) && (x0 + 1 < 27);
      bool ya = (y0 >= 0) && (y0 < 27), yb = (y0 + 1 >= 0) && (y0 + 1 < 27);
      if (xa && ya) v00 = cpb[(size_t)(y0 * 27 + x0) * D_MODEL + c];
      if (xb && ya) v10 = cpb[(size_t)(y0 * 27 + x0 + 1) * D_MODEL + c];
      if (xa && yb) v01 = cpb[(size_t)((y0 + 1) * 27 + x0) * D_MODEL + c];
      if (xb && yb) v11 = cpb[(size_t)((y0 + 1) * 27 + x0 + 1) * D_MODEL + c];
      acc += aw[h * 4 + p] *
             (v00 * wx0 * wy0 + v10 * wx1 * wy0 + v01 * wx0 * wy1 + v11 * wx1 * wy1);
    }
    sout[bq * D_MODEL + c] = acc;
  }
}

// ---------------------------------------------------------------------------
// Gate: gate[row] = sigmoid(dot(hg[row,96], W_g2) + b_g2). One wave per row.
// ---------------------------------------------------------------------------
__global__ __launch_bounds__(256) void gate_kernel(
    const float* __restrict__ hg, const float* __restrict__ w2,
    const float* __restrict__ b2, float* __restrict__ gate_ws,
    float* __restrict__ gate_out) {
  const int lane = threadIdx.x & 31;
  const int row  = blockIdx.x * 8 + (threadIdx.x >> 5);
  float s = 0.f;
  for (int k = lane; k < 96; k += 32) s += hg[(size_t)row * 96 + k] * w2[k];
#pragma unroll
  for (int off = 16; off > 0; off >>= 1) s += __shfl_xor(s, off, 32);
  if (lane == 0) {
    float gv = 1.0f / (1.0f + expf(-(s + b2[0])));
    gate_ws[row]  = gv;
    gate_out[row] = gv;
  }
}

// ---------------------------------------------------------------------------
// Final: fused = LN(vT + gate*f2) per token, written transposed to [B,384,64,64].
// ---------------------------------------------------------------------------
__global__ __launch_bounds__(256) void final_kernel(
    const float* __restrict__ vT, const float* __restrict__ f2,
    const float* __restrict__ gate, const float* __restrict__ g,
    const float* __restrict__ b, float* __restrict__ out) {
  __shared__ float S[D_MODEL][33];
  __shared__ float redS[32][8], redQ[32][8];
  __shared__ float stat[32][2];
  const int tid = threadIdx.x;
  const int bq0 = blockIdx.x * 32;
  const int bb  = blockIdx.y;

  for (int idx = tid; idx < 32 * D_MODEL; idx += 256) {
    int c = idx % D_MODEL, q = idx / D_MODEL;
    size_t o = ((size_t)bb * NQ + bq0 + q) * D_MODEL + c;
    S[c][q] = vT[o] + gate[(size_t)bb * NQ + bq0 + q] * f2[o];
  }
  __syncthreads();

  const int ql = tid >> 3, part = tid & 7;
  float s = 0.f, ss = 0.f;
  for (int c = part; c < D_MODEL; c += 8) {
    float v = S[c][ql];
    s += v; ss += v * v;
  }
  redS[ql][part] = s; redQ[ql][part] = ss;
  __syncthreads();
  if (part == 0) {
    float ts = 0.f, tq = 0.f;
#pragma unroll
    for (int p = 0; p < 8; ++p) { ts += redS[ql][p]; tq += redQ[ql][p]; }
    float mean = ts / D_MODEL;
    float var  = tq / D_MODEL - mean * mean;
    stat[ql][0] = mean;
    stat[ql][1] = rsqrtf(var + 1e-5f);
  }
  __syncthreads();

  for (int idx = tid; idx < 32 * D_MODEL; idx += 256) {
    int q = idx & 31, c = idx >> 5;
    float v = (S[c][q] - stat[q][0]) * stat[q][1] * g[c] + b[c];
    out[(size_t)bb * D_MODEL * NQ + (size_t)c * NQ + bq0 + q] = v;
  }
}

// ---------------------------------------------------------------------------
// Host-side launch
// ---------------------------------------------------------------------------
extern "C" void kernel_launch(void* const* d_in, const int* in_sizes, int n_in,
                              void* d_out, int out_size, void* d_ws, size_t ws_size,
                              hipStream_t stream) {
  (void)in_sizes; (void)n_in; (void)out_size; (void)ws_size;
  const float* vis    = (const float*)d_in[0];
  const float* cut3r  = (const float*)d_in[1];
  const float* ln_v_g = (const float*)d_in[2];
  const float* ln_v_b = (const float*)d_in[3];
  const float* ln_c_g = (const float*)d_in[4];
  const float* ln_c_b = (const float*)d_in[5];
  const float* W_c1   = (const float*)d_in[6];
  const float* b_c1   = (const float*)d_in[7];
  const float* W_c2   = (const float*)d_in[8];
  const float* b_c2   = (const float*)d_in[9];
  const float* W_off  = (const float*)d_in[10];
  const float* b_off  = (const float*)d_in[11];
  const float* W_attn = (const float*)d_in[12];
  const float* b_attn = (const float*)d_in[13];
  const float* W_out  = (const float*)d_in[14];
  const float* b_out  = (const float*)d_in[15];
  const float* W_f1   = (const float*)d_in[16];
  const float* b_f1   = (const float*)d_in[17];
  const float* W_f2   = (const float*)d_in[18];
  const float* b_f2   = (const float*)d_in[19];
  const float* W_g1   = (const float*)d_in[20];
  const float* b_g1   = (const float*)d_in[21];
  const float* W_g2   = (const float*)d_in[22];
  const float* b_g2   = (const float*)d_in[23];
  const float* ln_o_g = (const float*)d_in[24];
  const float* ln_o_b = (const float*)d_in[25];
  float* out = (float*)d_out;
  float* ws  = (float*)d_ws;

  // workspace layout (floats); aliases noted. Total ~444 MiB.
  size_t o = 0;
  float* vT   = ws + o; o += (size_t)BATCH * NQ * D_MODEL;   // 12.58M
  float* vn   = ws + o; o += (size_t)BATCH * NQ * D_MODEL;   // 12.58M
  float* cn   = ws + o; o += (size_t)BATCH * NCUT * DCUT;    // 5.98M
  float* h1   = ws + o; o += (size_t)BATCH * NCUT * 768;     // 4.49M
  float* cp   = ws + o; o += (size_t)BATCH * NCUT * D_MODEL; // 2.24M
  float* offr = ws + o; o += (size_t)BATCH * NQ * 64;        // 2.10M
  float* attr = ws + o; o += (size_t)BATCH * NQ * 32;        // 1.05M
  float* souT = ws + o; o += (size_t)BATCH * NQ * D_MODEL;   // 12.58M
  float* aout = ws + o; o += (size_t)BATCH * NQ * D_MODEL;   // 12.58M
  float* hf   = ws + o; o += (size_t)BATCH * NQ * 1536;      // 50.33M
  float* f2    = souT;  // alias: sout dead after W_out GEMM
  float* hg    = offr;  // alias: offr+attr (3.15M floats) dead after sampling
  float* gatev = h1;    // alias: h1 dead after cp GEMM

  const int Mv = BATCH * NQ;    // 32768
  const int Mc = BATCH * NCUT;  // 5840

  // 1) LN(cut3r) -> cn
  ln_rows_kernel<<<Mc, 256, 0, stream>>>(cut3r, ln_c_g, ln_c_b, cn, DCUT);
  // 2) h1 = gelu(cn @ W_c1 + b_c1)   [5840,768]
  gemm_wmma<1><<<dim3(6, 46), 256, 0, stream>>>(cn, W_c1, b_c1, h1, Mc, 768, DCUT);
  // 3) cp = h1 @ W_c2 + b_c2         [5840,384]
  gemm_wmma<0><<<dim3(3, 46), 256, 0, stream>>>(h1, W_c2, b_c2, cp, Mc, D_MODEL, 768);
  // 4) visual LN + transpose -> vT, vn
  ln_visual_kernel<<<dim3(128, 8), 256, 0, stream>>>(vis, ln_v_g, ln_v_b, vT, vn);
  // 5) offset projection  [32768,64]
  gemm_wmma<0><<<dim3(1, 256), 256, 0, stream>>>(vn, W_off, b_off, offr, Mv, 64, D_MODEL);
  // 6) attention projection [32768,32]
  gemm_wmma<0><<<dim3(1, 256), 256, 0, stream>>>(vn, W_attn, b_attn, attr, Mv, 32, D_MODEL);
  // 7) deformable sampling -> sout [32768,384]
  sample_kernel<<<dim3(NQ, BATCH), 128, 0, stream>>>(offr, attr, cp, souT);
  // 8) attn_out = sout @ W_out + b_out
  gemm_wmma<0><<<dim3(3, 256), 256, 0, stream>>>(souT, W_out, b_out, aout, Mv, D_MODEL, D_MODEL);
  // 9) hf = gelu(attn_out @ W_f1 + b_f1)  [32768,1536]
  gemm_wmma<1><<<dim3(12, 256), 256, 0, stream>>>(aout, W_f1, b_f1, hf, Mv, 1536, D_MODEL);
  // 10) f2 = hf @ W_f2 + b_f2            [32768,384]
  gemm_wmma<0><<<dim3(3, 256), 256, 0, stream>>>(hf, W_f2, b_f2, f2, Mv, D_MODEL, 1536);
  // 11) gate hidden: relu(vT @ W_g1 + b_g1)  [32768,96]
  gemm_wmma<2><<<dim3(1, 256), 256, 0, stream>>>(vT, W_g1, b_g1, hg, Mv, 96, D_MODEL);
  // 12) gate = sigmoid(hg @ W_g2 + b_g2); also written to output tail
  gate_kernel<<<Mv / 8, 256, 0, stream>>>(hg, W_g2, b_g2, gatev,
                                          out + (size_t)BATCH * D_MODEL * NQ);
  // 13) fused = LN(vT + gate*f2), transposed to [B,384,64,64]
  final_kernel<<<dim3(128, 8), 256, 0, stream>>>(vT, f2, gatev, ln_o_g, ln_o_b, out);
}
